// AttentionLoRAText_35278861370035
// MI455X (gfx1250) — matrix operand order repaired and checked by reading, not compiled, not run on hardware
//
#include <hip/hip_runtime.h>

typedef __attribute__((ext_vector_type(16))) __bf16       bf16x16;
typedef __attribute__((ext_vector_type(8)))  float        f32x8;
typedef __attribute__((ext_vector_type(4)))  float        f32x4;
typedef __attribute__((ext_vector_type(4)))  unsigned int u32x4;

constexpr int NB  = 4;     // batch
constexpr int SEQ = 1024;  // sequence length
constexpr int DM  = 2048;  // model dim
constexpr int NH  = 16;    // heads
constexpr int DH  = 128;   // head dim

union Frag {
  u32x4   u[2];
  bf16x16 v;
};

__device__ __forceinline__ unsigned short f2bf(float f) {
  unsigned b = __float_as_uint(f);
  b += 0x7FFFu + ((b >> 16) & 1u);   // round-to-nearest-even
  return (unsigned short)(b >> 16);
}
__device__ __forceinline__ unsigned pack2(float lo, float hi) {
  return (unsigned)f2bf(lo) | ((unsigned)f2bf(hi) << 16);
}
__device__ __forceinline__ unsigned lds_off(const void* p) {
  return (unsigned)(size_t)p;        // LDS aperture: DS offset = low 32 bits
}
__device__ __forceinline__ void async_copy16(unsigned lds_byte, const void* gsrc) {
  asm volatile("global_load_async_to_lds_b128 %0, %1, off"
               :: "v"(lds_byte), "v"((unsigned long long)(size_t)gsrc)
               : "memory");
}
__device__ __forceinline__ void wait_async0() {
#if __has_builtin(__builtin_amdgcn_s_wait_asynccnt)
  __builtin_amdgcn_s_wait_asynccnt(0);
#else
  asm volatile("s_wait_asynccnt 0" ::: "memory");
#endif
}

// ---------------------------------------------------------------------------
// One-shot fp32 -> bf16 conversion (8 elements per thread).
// ---------------------------------------------------------------------------
__global__ void __launch_bounds__(256)
cvt_kernel(const float* __restrict__ src, unsigned short* __restrict__ dst) {
  size_t base = ((size_t)blockIdx.x * 256 + threadIdx.x) * 8;
  f32x4 a = *(const f32x4*)(src + base);
  f32x4 b = *(const f32x4*)(src + base + 4);
  u32x4 p;
  p.x = pack2(a.x, a.y); p.y = pack2(a.z, a.w);
  p.z = pack2(b.x, b.y); p.w = pack2(b.z, b.w);
  *(u32x4*)(dst + base) = p;
}

// ---------------------------------------------------------------------------
// GEMM: C(4096x2048) = A(bf16) @ W^T(bf16) + bias.
// Double-buffered LDS staging via global_load_async_to_lds_b128 (ASYNCcnt);
// steady-state loop body is branch-free: wait -> barrier -> stage k+1 ->
// compute k.  Last iteration peeled so no conditional lives in the loop.
// Block 256 thr (8 waves), tile 128x128, BK=32, v_wmma_f32_16x16x32_bf16.
// mode 0: f32 to outF row-major                               [output proj]
// mode 1: RoPE + 1/sqrt(128), bf16 to outB (B,H,S,HD)         [Q]
// mode 2: RoPE,               bf16 to outB (B,H,S,HD)         [K]
// mode 3: bf16 to outB (B,H,HD,S)  (V transposed)             [V]
// ---------------------------------------------------------------------------
__global__ void __launch_bounds__(256)
gemm_kernel(const unsigned short* __restrict__ A, const unsigned short* __restrict__ W,
            const float* __restrict__ bias, const float* __restrict__ fc,
            const float* __restrict__ fs, float* __restrict__ outF,
            unsigned short* __restrict__ outB, int mode) {
  constexpr int LS = 40;  // ushorts per LDS row: 32 data + 8 pad (80B, 16B-mult)
  constexpr int NT = DM / 32;
  __shared__ unsigned short lds[2][2][128 * LS];  // [parity][A/B], 40 KB

  const int tid  = threadIdx.x;
  const int w    = tid >> 5, lane = tid & 31;
  const int wm   = w >> 2,   wn   = w & 3;     // 2x4 wave grid, 64x32 each
  const int hs   = lane >> 4, lr  = lane & 15;
  const int m0   = blockIdx.y * 128;
  const int n0   = blockIdx.x * 128;

  f32x8 zero = {};
  f32x8 acc[4][2];
#pragma unroll
  for (int i = 0; i < 4; i++)
#pragma unroll
    for (int j = 0; j < 2; j++) acc[i][j] = zero;

  // stage tile (parity p, k-offset k0): 2 x b128 per thread per matrix
#define STAGE(p, k0)                                                          \
  do {                                                                        \
    _Pragma("unroll")                                                         \
    for (int i_ = 0; i_ < 2; i_++) {                                          \
      int idx_ = tid + i_ * 256;                                              \
      int row_ = idx_ >> 2;                                                   \
      int c8_  = (idx_ & 3) * 8;                                              \
      async_copy16(lds_off(&lds[p][0][row_ * LS + c8_]),                      \
                   A + (size_t)(m0 + row_) * DM + (k0) + c8_);                \
      async_copy16(lds_off(&lds[p][1][row_ * LS + c8_]),                      \
                   W + (size_t)(n0 + row_) * DM + (k0) + c8_);                \
    }                                                                         \
  } while (0)

  // compute on parity p: frag ds_loads + 8 WMMA
#define COMPUTE(p)                                                            \
  do {                                                                        \
    const unsigned short* tA_ = lds[p][0];                                    \
    const unsigned short* tB_ = lds[p][1];                                    \
    Frag a_[4], b_[2];                                                        \
    _Pragma("unroll")                                                         \
    for (int mt_ = 0; mt_ < 4; mt_++) {  /* A frag: chunks hs*8, hs*8+16 */   \
      int base_ = (wm * 64 + mt_ * 16 + lr) * LS + hs * 8;                    \
      a_[mt_].u[0] = *(const u32x4*)&tA_[base_];                              \
      a_[mt_].u[1] = *(const u32x4*)&tA_[base_ + 16];                         \
    }                                                                         \
    _Pragma("unroll")                                                         \
    for (int nt_ = 0; nt_ < 2; nt_++) {  /* B frag: chunks hs*8, hs*8+16 */   \
      int base_ = (wn * 32 + nt_ * 16 + lr) * LS + hs * 8;                    \
      b_[nt_].u[0] = *(const u32x4*)&tB_[base_];                              \
      b_[nt_].u[1] = *(const u32x4*)&tB_[base_ + 16];                         \
    }                                                                         \
    _Pragma("unroll")                                                         \
    for (int mt_ = 0; mt_ < 4; mt_++)                                         \
      _Pragma("unroll")                                                       \
      for (int nt_ = 0; nt_ < 2; nt_++)                                       \
        acc[mt_][nt_] = __builtin_amdgcn_wmma_f32_16x16x32_bf16(              \
            false, a_[mt_].v, false, b_[nt_].v, (short)0, acc[mt_][nt_],      \
            false, false);                                                    \
  } while (0)

  STAGE(0, 0);                                 // prologue: prefetch tile 0
  for (int kt = 0; kt < NT - 1; kt++) {        // branch-free steady state
    wait_async0();                             // tile kt landed (this wave)
    __syncthreads();                           // visible to whole block
    STAGE((kt + 1) & 1, (kt + 1) * 32);        // DMA next tile during compute
    COMPUTE(kt & 1);
  }
  wait_async0();                               // peeled last iteration
  __syncthreads();
  COMPUTE((NT - 1) & 1);
#undef STAGE
#undef COMPUTE

  const float scaleQ = 0.08838834764831845f;  // 1/sqrt(128)
#pragma unroll
  for (int mt = 0; mt < 4; mt++) {
#pragma unroll
    for (int nt = 0; nt < 2; nt++) {
#pragma unroll
      for (int r = 0; r < 8; r++) {
        float v  = acc[mt][nt][r];
        int   gm = m0 + wm * 64 + mt * 16 + hs * 8 + r;  // C: m = r + 8*hs
        int   gn = n0 + wn * 32 + nt * 16 + lr;          // C: n = lane&15
        v += bias[gn];
        if (mode == 0) {
          outF[(size_t)gm * DM + gn] = v;
        } else {
          int bb = gm >> 10;        // / SEQ
          int s  = gm & (SEQ - 1);
          int h  = gn >> 7;         // / DH
          int hd = gn & (DH - 1);
          if (mode == 3) {          // V transposed: (B,H,HD,S)
            outB[((size_t)(bb * NH + h) * DH + hd) * SEQ + s] = f2bf(v);
          } else {                  // RoPE: adjacent columns = adjacent lanes
            float pv = __shfl_xor(v, 1, 32);
            float c  = fc[s * (DH / 2) + (hd >> 1)];
            float sn = fs[s * (DH / 2) + (hd >> 1)];
            float o  = ((lane & 1) == 0) ? (v * c - pv * sn) : (pv * sn + v * c);
            if (mode == 1) o *= scaleQ;
            outB[((size_t)(bb * NH + h) * SEQ + s) * DH + hd] = f2bf(o);
          }
        }
      }
    }
  }
}

// ---------------------------------------------------------------------------
// Flash attention. Grid (S/128, B*H), 256 thr (8 waves).
// Wave owns 16 Q rows; streams causal key blocks of 32; online softmax;
// O in 8 f32 WMMA tiles; writes bf16 attn in (B*S, DM) layout.
// ---------------------------------------------------------------------------
__global__ void __launch_bounds__(256)
attn_kernel(const unsigned short* __restrict__ Q,
            const unsigned short* __restrict__ K,
            const unsigned short* __restrict__ VT,
            unsigned short* __restrict__ O) {
  __shared__ unsigned short lP[8 * 16 * 40];  // per-wave 16x32 P tile (pad 40)
  const int tid = threadIdx.x;
  const int w   = tid >> 5, lane = tid & 31;
  const int hs  = lane >> 4, lr  = lane & 15;
  const int bh  = blockIdx.y;                 // b*NH + h
  const int q0  = blockIdx.x * 128 + w * 16;

  const unsigned short* qp = Q  + (size_t)bh * SEQ * DH;
  const unsigned short* kp = K  + (size_t)bh * SEQ * DH;
  const unsigned short* vp = VT + (size_t)bh * DH * SEQ;
  unsigned short* myP = lP + w * 16 * 40;

  Frag aQ[4];
#pragma unroll
  for (int kk = 0; kk < 4; kk++) {            // Q pre-scaled by 1/sqrt(DH)
    const unsigned short* p = qp + (size_t)(q0 + lr) * DH + kk * 32 + hs * 8;
    aQ[kk].u[0] = *(const u32x4*)p;
    aQ[kk].u[1] = *(const u32x4*)(p + 16);
  }

  f32x8 zero = {};
  f32x8 accO[8];
#pragma unroll
  for (int j = 0; j < 8; j++) accO[j] = zero;
  float m_i[8], l_i[8];
#pragma unroll
  for (int r = 0; r < 8; r++) { m_i[r] = -1e30f; l_i[r] = 0.f; }

  const int nkb = (q0 + 47) >> 5;             // causal trip count (wave-uniform)
  for (int kb = 0; kb < nkb; kb++) {
    const int kbase = kb * 32;
    f32x8 sa0 = zero, sa1 = zero;             // S tile 16x32 (two 16x16)
#pragma unroll
    for (int kk = 0; kk < 4; kk++) {
      Frag b0, b1;                            // B^T rows = K rows (contig hd)
      const unsigned short* p0 = kp + (size_t)(kbase + lr) * DH + kk * 32 + hs * 8;
      b0.u[0] = *(const u32x4*)p0;  b0.u[1] = *(const u32x4*)(p0 + 16);
      const unsigned short* p1 = kp + (size_t)(kbase + 16 + lr) * DH + kk * 32 + hs * 8;
      b1.u[0] = *(const u32x4*)p1;  b1.u[1] = *(const u32x4*)(p1 + 16);
      sa0 = __builtin_amdgcn_wmma_f32_16x16x32_bf16(false, aQ[kk].v, false, b0.v,
                                                    (short)0, sa0, false, false);
      sa1 = __builtin_amdgcn_wmma_f32_16x16x32_bf16(false, aQ[kk].v, false, b1.v,
                                                    (short)0, sa1, false, false);
    }
#pragma unroll
    for (int r = 0; r < 8; r++) {             // online softmax per Q row
      int   qrow = q0 + hs * 8 + r;
      float s0 = sa0[r], s1 = sa1[r];
      s0 = (kbase + lr > qrow)      ? -1e30f : s0;   // causal mask, branchless
      s1 = (kbase + 16 + lr > qrow) ? -1e30f : s1;
      float mx = fmaxf(s0, s1);
      mx = fmaxf(mx, __shfl_xor(mx, 1, 32));
      mx = fmaxf(mx, __shfl_xor(mx, 2, 32));
      mx = fmaxf(mx, __shfl_xor(mx, 4, 32));
      mx = fmaxf(mx, __shfl_xor(mx, 8, 32));
      float mnew = fmaxf(m_i[r], mx);
      float sc   = __expf(m_i[r] - mnew);
      float p0   = __expf(s0 - mnew);
      float p1   = __expf(s1 - mnew);
      float rs = p0 + p1;
      rs += __shfl_xor(rs, 1, 32);
      rs += __shfl_xor(rs, 2, 32);
      rs += __shfl_xor(rs, 4, 32);
      rs += __shfl_xor(rs, 8, 32);
      l_i[r] = l_i[r] * sc + rs;
      m_i[r] = mnew;
#pragma unroll
      for (int j = 0; j < 8; j++) accO[j][r] *= sc;
      int pm = hs * 8 + r;                    // stage P (C-layout -> LDS rows)
      myP[pm * 40 + lr]      = f2bf(p0);
      myP[pm * 40 + 16 + lr] = f2bf(p1);
    }
    asm volatile("s_wait_dscnt 0" ::: "memory");  // wave-local LDS st->ld
    Frag aP;
    {
      const unsigned short* p = myP + lr * 40 + hs * 8;  // A-frag pattern
      aP.u[0] = *(const u32x4*)p;
      aP.u[1] = *(const u32x4*)(p + 16);
    }
#pragma unroll
    for (int j = 0; j < 8; j++) {             // O += P @ V, B^T rows = VT rows
      Frag bV;
      const unsigned short* p = vp + (size_t)(j * 16 + lr) * SEQ + kbase + hs * 8;
      bV.u[0] = *(const u32x4*)p;
      bV.u[1] = *(const u32x4*)(p + 16);
      accO[j] = __builtin_amdgcn_wmma_f32_16x16x32_bf16(false, aP.v, false, bV.v,
                                                        (short)0, accO[j], false, false);
    }
  }

  float inv[8];
#pragma unroll
  for (int r = 0; r < 8; r++) inv[r] = 1.0f / l_i[r];
  const int bb = bh >> 4, h = bh & 15;
#pragma unroll
  for (int j = 0; j < 8; j++)
#pragma unroll
    for (int r = 0; r < 8; r++) {
      int qrow = q0 + hs * 8 + r;
      O[(size_t)(bb * SEQ + qrow) * DM + h * DH + j * 16 + lr] =
          f2bf(accO[j][r] * inv[r]);
    }
}

// ---------------------------------------------------------------------------
extern "C" void kernel_launch(void* const* d_in, const int* in_sizes, int n_in,
                              void* d_out, int out_size, void* d_ws, size_t ws_size,
                              hipStream_t stream) {
  (void)in_sizes; (void)n_in; (void)out_size; (void)ws_size;
  const float* x  = (const float*)d_in[0];
  const float* fc = (const float*)d_in[2];
  const float* fs = (const float*)d_in[3];
  const float* wq = (const float*)d_in[5];
  const float* bq = (const float*)d_in[6];
  const float* wk = (const float*)d_in[7];
  const float* bk = (const float*)d_in[8];
  const float* wv = (const float*)d_in[9];
  const float* bv = (const float*)d_in[10];
  const float* wo = (const float*)d_in[11];
  const float* bo = (const float*)d_in[12];
  float* out = (float*)d_out;

  const size_t NX = (size_t)NB * SEQ * DM;  // 8M elems (x / qkv / attn)
  const size_t NW = (size_t)DM * DM;        // 4M elems (weights)
  unsigned short* xb   = (unsigned short*)d_ws;
  unsigned short* wqb  = xb  + NX;
  unsigned short* wkb  = wqb + NW;
  unsigned short* wvb  = wkb + NW;
  unsigned short* wob  = wvb + NW;
  unsigned short* qb   = wob + NW;
  unsigned short* kbf  = qb  + NX;
  unsigned short* vtb  = kbf + NX;
  unsigned short* attn = vtb + NX;          // total 56M ushorts = 112 MB

  cvt_kernel<<<NX / (8 * 256), 256, 0, stream>>>(x,  xb);
  cvt_kernel<<<NW / (8 * 256), 256, 0, stream>>>(wq, wqb);
  cvt_kernel<<<NW / (8 * 256), 256, 0, stream>>>(wk, wkb);
  cvt_kernel<<<NW / (8 * 256), 256, 0, stream>>>(wv, wvb);
  cvt_kernel<<<NW / (8 * 256), 256, 0, stream>>>(wo, wob);

  dim3 gg(DM / 128, (NB * SEQ) / 128);      // 16 x 32
  gemm_kernel<<<gg, 256, 0, stream>>>(xb, wqb, bq, fc, fs, nullptr, qb, 1);
  gemm_kernel<<<gg, 256, 0, stream>>>(xb, wkb, bk, fc, fs, nullptr, kbf, 2);
  gemm_kernel<<<gg, 256, 0, stream>>>(xb, wvb, bv, fc, fs, nullptr, vtb, 3);
  dim3 ga(SEQ / 128, NB * NH);              // 8 x 64
  attn_kernel<<<ga, 256, 0, stream>>>(qb, kbf, vtb, attn);
  gemm_kernel<<<gg, 256, 0, stream>>>(attn, wob, bo, fc, fs, out, nullptr, 0);
}
